// AttentionBlock_22351009809168
// MI455X (gfx1250) — matrix operand order repaired and checked by reading, not compiled, not run on hardware
//
#include <hip/hip_runtime.h>
#include <hip/hip_bf16.h>

// ---------------------------------------------------------------------------
// AttentionBlock for MI455X (gfx1250, wave32, WMMA f16 -> f32 accumulate)
//   x:(16,512,32,32) f32 -> GN(8) -> qkv GEMM -> QK^T -> softmax -> AV -> proj
// NT GEMMs: 128x64 block tile, BK=32, 8 waves/block, 32x32 wave tile
// (4x v_wmma_f32_16x16x32_f16 per wave per K step), async-to-LDS staging
// (global_load_async_to_lds_b128, ASYNCcnt) with manual double buffering.
// Stage loop is fully unrolled so buffer selects/offsets are compile-time and
// accumulators stay in place (no v_mov shuffles).
// ---------------------------------------------------------------------------

typedef __attribute__((ext_vector_type(16))) _Float16     v16h;
typedef __attribute__((ext_vector_type(8)))  float        v8f;
typedef __attribute__((ext_vector_type(4)))  unsigned int u32x4;

#define Bn   16
#define Cc   512
#define Nn   1024
#define O3   1536
#define GRP  8
#define CPG  64
#define LDST 40          // LDS tile row stride in halfs (32 + 8 pad)
#define SCALE 0.044194173824159216f   // 1/sqrt(512)

#define WMMA_F16(a, b, c)                                                     \
  __builtin_amdgcn_wmma_f32_16x16x32_f16(false, (a), false, (b), (short)0,    \
                                         (c), false, false)

__device__ __forceinline__ void copy16B(void* d, const void* s) {
  *(u32x4*)d = *(const u32x4*)s;
}

// Async L2->LDS 16-byte copy (gfx1250). VDST VGPR holds the LDS byte address
// (generic-pointer low 32 bits == LDS offset per the flat-aperture rule);
// tracked with ASYNCcnt.
__device__ __forceinline__ void async_b128(unsigned ldsOff, const _Float16* g) {
  asm volatile("global_load_async_to_lds_b128 %0, %1, off"
               :: "v"(ldsOff), "v"(g) : "memory");
}

// A fragment: 16x32 f16, ISA layout: lanes 0-15 row=lane K={kb..kb+7,kb+16..kb+23}
// with kb = 8*(lane>>4). Rows contiguous in LDS (stride LDST halfs).
__device__ __forceinline__ v16h frag_a(const _Float16* sA, int lane) {
  const _Float16* p = sA + (lane & 15) * LDST + ((lane >> 4) << 3);
  union { u32x4 q[2]; v16h v; } u;
  u.q[0] = *(const u32x4*)p;          // K kb..kb+7
  u.q[1] = *(const u32x4*)(p + 16);   // K kb+16..kb+23
  return u.v;
}

// B fragment: 32x16 f16 (KxN), stored transposed in LDS as Bt[n][k]:
// lane col = colBase + (lane&15), K = {kb..kb+7, kb+16..kb+23}, kb = 8*(lane>>4).
__device__ __forceinline__ v16h frag_b(const _Float16* sBt, int lane, int colBase) {
  const _Float16* p = sBt + (colBase + (lane & 15)) * LDST + ((lane >> 4) << 3);
  union { u32x4 q[2]; v16h v; } u;
  u.q[0] = *(const u32x4*)p;
  u.q[1] = *(const u32x4*)(p + 16);
  return u.v;
}

// ---------------------------------------------------------------------------
// Shared NT GEMM body: C(128x64) = A(128xK, ldA) * Bt(64xK, ldB)^T
// Double-buffered async-to-LDS staging; 4 accumulators per wave (32x32 tile).
// Fully unrolled: all buffer selects and K offsets are compile-time.
// ---------------------------------------------------------------------------
template <int KDIM>
__device__ __forceinline__ void gemm_nt_async(const _Float16* __restrict__ A, int ldA,
                                              const _Float16* __restrict__ Bt, int ldB,
                                              _Float16* sA, _Float16* sB, int tid,
                                              v8f& a00, v8f& a01, v8f& a10, v8f& a11) {
  const int lane = tid & 31, wave = tid >> 5;
  const int wm = wave >> 1, wn = wave & 1;
  const int r = tid >> 2, o = (tid & 3) << 3;
  constexpr int S = KDIM / 32;
  constexpr unsigned bA = 128u * LDST * sizeof(_Float16);  // A buffer stride (bytes)
  constexpr unsigned bB = 64u * LDST * sizeof(_Float16);   // B buffer stride (bytes)
  constexpr int bufAh = 128 * LDST;                        // halfs
  constexpr int bufBh = 64 * LDST;
  a00 = {}; a01 = {}; a10 = {}; a11 = {};

  const _Float16* gA0 = A + (size_t)r * ldA + o;
  const _Float16* gA1 = A + (size_t)(r + 64) * ldA + o;
  const _Float16* gB  = Bt + (size_t)r * ldB + o;
  const unsigned dA0 = (unsigned)(uintptr_t)(sA + r * LDST + o);
  const unsigned dA1 = dA0 + 64u * LDST * sizeof(_Float16);
  const unsigned dB  = (unsigned)(uintptr_t)(sB + r * LDST + o);

  // prologue: stage 0 -> buffer 0 (3 async loads per thread, in-order done)
  async_b128(dA0, gA0);
  async_b128(dA1, gA1);
  async_b128(dB,  gB);

#pragma unroll
  for (int i = 0; i < S; ++i) {
    if (i + 1 < S) {
      const unsigned nb = ((i + 1) & 1) ? 1u : 0u;   // compile-time after unroll
      const int nk = 32 * (i + 1);
      async_b128(dA0 + nb * bA, gA0 + nk);
      async_b128(dA1 + nb * bA, gA1 + nk);
      async_b128(dB  + nb * bB, gB  + nk);
      // async loads complete in order: <=3 outstanding => stage i has landed
      asm volatile("s_wait_asynccnt 0x3" ::: "memory");
    } else {
      asm volatile("s_wait_asynccnt 0x0" ::: "memory");
    }
    __syncthreads();
    const _Float16* cA = sA + (i & 1) * bufAh + wm * 32 * LDST;
    const _Float16* cB = sB + (i & 1) * bufBh;
    v16h fa0 = frag_a(cA, lane);
    v16h fa1 = frag_a(cA + 16 * LDST, lane);
    v16h fb0 = frag_b(cB, lane, wn * 32);
    v16h fb1 = frag_b(cB, lane, wn * 32 + 16);
    a00 = WMMA_F16(fa0, fb0, a00);
    a01 = WMMA_F16(fa0, fb1, a01);
    a10 = WMMA_F16(fa1, fb0, a10);
    a11 = WMMA_F16(fa1, fb1, a11);
    __syncthreads();   // all reads done before this buffer is refilled
  }
}

__device__ __forceinline__ float block_sum(float v, float* red, int tid) {
  red[tid] = v; __syncthreads();
#pragma unroll
  for (int s = 128; s > 0; s >>= 1) {
    if (tid < s) red[tid] += red[tid + s];
    __syncthreads();
  }
  float r = red[0]; __syncthreads();
  return r;
}

__device__ __forceinline__ float block_max(float v, float* red, int tid) {
  red[tid] = v; __syncthreads();
#pragma unroll
  for (int s = 128; s > 0; s >>= 1) {
    if (tid < s) red[tid] = fmaxf(red[tid], red[tid + s]);
    __syncthreads();
  }
  float r = red[0]; __syncthreads();
  return r;
}

// ---------------------------------------------------------------------------
// f32 -> f16 weight conversion
// ---------------------------------------------------------------------------
__global__ void k_cvt(const float* __restrict__ s, _Float16* __restrict__ d, int n) {
  int i = blockIdx.x * 256 + threadIdx.x;
  if (i < n) d[i] = (_Float16)s[i];
}

// ---------------------------------------------------------------------------
// GroupNorm over (C/G,H,W) per (b,g); writes h token-major (B,N,C) as f16.
// ---------------------------------------------------------------------------
__global__ __launch_bounds__(256) void k_gn(const float* __restrict__ x,
                                            const float* __restrict__ gw,
                                            const float* __restrict__ gb,
                                            _Float16* __restrict__ h16) {
  __shared__ float red[256];
  const int b = blockIdx.x >> 3, g = blockIdx.x & 7;
  const int tid = threadIdx.x;
  const float* xp = x + ((size_t)b * Cc + g * CPG) * Nn;
  float s = 0.f, q = 0.f;
  for (int i = tid; i < CPG * Nn; i += 256) { float v = xp[i]; s += v; q += v * v; }
  s = block_sum(s, red, tid);
  q = block_sum(q, red, tid);
  const float inv  = 1.0f / (float)(CPG * Nn);
  const float mean = s * inv;
  const float var  = q * inv - mean * mean;
  const float rstd = rsqrtf(var + 1e-5f);
  for (int i = tid; i < CPG * Nn; i += 256) {
    const int c = i >> 10, n = i & 1023;
    const int ch = g * CPG + c;
    const float v = (xp[i] - mean) * rstd * gw[ch] + gb[ch];
    h16[((size_t)b * Nn + n) * Cc + ch] = (_Float16)v;
  }
}

// ---------------------------------------------------------------------------
// QKV GEMM: out[b,n,o] = sum_c h[b,n,c]*W[o,c] + bias[o]  (NT, f16 out)
// ---------------------------------------------------------------------------
__global__ __launch_bounds__(256) void k_qkv(const _Float16* __restrict__ h16,
                                             const _Float16* __restrict__ w16,
                                             const float* __restrict__ bias,
                                             _Float16* __restrict__ qkvo) {
  __shared__ _Float16 sA[2 * 128 * LDST];
  __shared__ _Float16 sB[2 * 64 * LDST];
  const int b = blockIdx.z, mBase = blockIdx.x * 128, nBase = blockIdx.y * 64;
  const int tid = threadIdx.x, lane = tid & 31, wave = tid >> 5;
  const int wm = wave >> 1, wn = wave & 1;
  v8f a00, a01, a10, a11;
  gemm_nt_async<Cc>(h16 + ((size_t)b * Nn + mBase) * Cc, Cc,
                    w16 + (size_t)nBase * Cc, Cc, sA, sB, tid, a00, a01, a10, a11);
  const int row0 = mBase + wm * 32 + ((lane >> 4) << 3);
  const int c0 = nBase + wn * 32 + (lane & 15), c1 = c0 + 16;
  const float b0f = bias[c0], b1f = bias[c1];
  _Float16* outp = qkvo + (size_t)b * Nn * O3;
#pragma unroll
  for (int r = 0; r < 8; ++r) {
    const size_t base0 = (size_t)(row0 + r) * O3;
    const size_t base1 = (size_t)(row0 + 16 + r) * O3;
    outp[base0 + c0] = (_Float16)(a00[r] + b0f);
    outp[base0 + c1] = (_Float16)(a01[r] + b1f);
    outp[base1 + c0] = (_Float16)(a10[r] + b0f);
    outp[base1 + c1] = (_Float16)(a11[r] + b1f);
  }
}

// ---------------------------------------------------------------------------
// Scores GEMM: scores[b,n,m] = (1/sqrt(C)) * sum_c q[b,n,c]*k[b,m,c]
// ---------------------------------------------------------------------------
__global__ __launch_bounds__(256) void k_scores(const _Float16* __restrict__ qkvo,
                                                float* __restrict__ scores) {
  __shared__ _Float16 sA[2 * 128 * LDST];
  __shared__ _Float16 sB[2 * 64 * LDST];
  const int b = blockIdx.z, mBase = blockIdx.x * 128, nBase = blockIdx.y * 64;
  const int tid = threadIdx.x, lane = tid & 31, wave = tid >> 5;
  const int wm = wave >> 1, wn = wave & 1;
  v8f a00, a01, a10, a11;
  gemm_nt_async<Cc>(qkvo + ((size_t)b * Nn + mBase) * O3, O3,         // q
                    qkvo + ((size_t)b * Nn + nBase) * O3 + Cc, O3,    // k
                    sA, sB, tid, a00, a01, a10, a11);
  const int row0 = mBase + wm * 32 + ((lane >> 4) << 3);
  const int c0 = nBase + wn * 32 + (lane & 15);
  float* p = scores + (size_t)b * Nn * Nn;
#pragma unroll
  for (int r = 0; r < 8; ++r) {
    const size_t base0 = (size_t)(row0 + r) * Nn;
    const size_t base1 = (size_t)(row0 + 16 + r) * Nn;
    p[base0 + c0]      = a00[r] * SCALE;
    p[base0 + c0 + 16] = a01[r] * SCALE;
    p[base1 + c0]      = a10[r] * SCALE;
    p[base1 + c0 + 16] = a11[r] * SCALE;
  }
}

// ---------------------------------------------------------------------------
// Softmax over each 1024-row; result written IN PLACE as f16 (row stride stays
// 1024 floats = 2048 halfs for the AV GEMM A operand).
// ---------------------------------------------------------------------------
__global__ __launch_bounds__(256) void k_softmax(float* __restrict__ scores) {
  __shared__ float red[256];
  const int tid = threadIdx.x;
  float* p = scores + (size_t)blockIdx.x * Nn;
  float v[4];
#pragma unroll
  for (int j = 0; j < 4; ++j) v[j] = p[tid + 256 * j];
  float m = fmaxf(fmaxf(v[0], v[1]), fmaxf(v[2], v[3]));
  m = block_max(m, red, tid);
  float e[4], s = 0.f;
#pragma unroll
  for (int j = 0; j < 4; ++j) { e[j] = __expf(v[j] - m); s += e[j]; }
  s = block_sum(s, red, tid);
  const float inv = 1.0f / s;
  _Float16* ph = (_Float16*)p;
  __syncthreads();   // all reads of this row complete before in-place f16 write
#pragma unroll
  for (int j = 0; j < 4; ++j) ph[tid + 256 * j] = (_Float16)(e[j] * inv);
}

// ---------------------------------------------------------------------------
// AV GEMM: h2[b,n,c] = sum_m attn[b,n,m] * v[b,m,c]
// A = attn (f16 in-place, lda = 2*Nn halfs). B is K-major (v rows are K) so
// the LDS tile is transposed on load (manual staging; async copy cannot
// transpose). f16 output, token-major (N,C). 128x64 block, 32x32 wave tile.
// ---------------------------------------------------------------------------
__global__ __launch_bounds__(256) void k_av(const float* __restrict__ scores,
                                            const _Float16* __restrict__ qkvo,
                                            _Float16* __restrict__ h2) {
  __shared__ _Float16 sA[128 * LDST];
  __shared__ _Float16 sB[64 * LDST];
  const int b = blockIdx.z, mBase = blockIdx.x * 128, nBase = blockIdx.y * 64;
  const int tid = threadIdx.x, lane = tid & 31, wave = tid >> 5;
  const int wm = wave >> 1, wn = wave & 1;
  const _Float16* A = (const _Float16*)(scores + ((size_t)b * Nn + mBase) * Nn);
  const _Float16* V = qkvo + (size_t)b * Nn * O3 + 2 * Cc;   // v[m][c] = V[m*O3+c]
  const int r4 = tid >> 2, o4 = (tid & 3) << 3;  // A loader
  const int kT = tid >> 3, nT = (tid & 7) << 3;  // B loader: k row, 8 n's/thread
  v8f a00 = {}, a01 = {}, a10 = {}, a11 = {};
  for (int k = 0; k < Nn; k += 32) {
    copy16B(sA + r4 * LDST + o4,        A + (size_t)r4 * (2 * Nn) + k + o4);
    copy16B(sA + (r4 + 64) * LDST + o4, A + (size_t)(r4 + 64) * (2 * Nn) + k + o4);
    {
      union { u32x4 q; _Float16 h[8]; } t;
      t.q = *(const u32x4*)(V + (size_t)(k + kT) * O3 + nBase + nT);
#pragma unroll
      for (int j = 0; j < 8; ++j) sB[(nT + j) * LDST + kT] = t.h[j];
    }
    if (k + 32 < Nn) {
      __builtin_prefetch(A + (size_t)r4 * (2 * Nn) + k + 32 + o4, 0, 1);
      __builtin_prefetch(A + (size_t)(r4 + 64) * (2 * Nn) + k + 32 + o4, 0, 1);
    }
    __syncthreads();
    const _Float16* cA = sA + wm * 32 * LDST;
    v16h fa0 = frag_a(cA, lane);
    v16h fa1 = frag_a(cA + 16 * LDST, lane);
    v16h fb0 = frag_b(sB, lane, wn * 32);
    v16h fb1 = frag_b(sB, lane, wn * 32 + 16);
    a00 = WMMA_F16(fa0, fb0, a00);
    a01 = WMMA_F16(fa0, fb1, a01);
    a10 = WMMA_F16(fa1, fb0, a10);
    a11 = WMMA_F16(fa1, fb1, a11);
    __syncthreads();
  }
  const int row0 = mBase + wm * 32 + ((lane >> 4) << 3);
  const int c0 = nBase + wn * 32 + (lane & 15);
#pragma unroll
  for (int r = 0; r < 8; ++r) {
    const size_t base0 = ((size_t)b * Nn + row0 + r) * Cc;
    const size_t base1 = ((size_t)b * Nn + row0 + 16 + r) * Cc;
    h2[base0 + c0]      = (_Float16)a00[r];
    h2[base0 + c0 + 16] = (_Float16)a01[r];
    h2[base1 + c0]      = (_Float16)a10[r];
    h2[base1 + c0 + 16] = (_Float16)a11[r];
  }
}

// ---------------------------------------------------------------------------
// Proj GEMM + bias + residual, transposed store to (B,C,N) f32 output.
// ---------------------------------------------------------------------------
__global__ __launch_bounds__(256) void k_proj(const _Float16* __restrict__ h2,
                                              const _Float16* __restrict__ w16,
                                              const float* __restrict__ bias,
                                              const float* __restrict__ x,
                                              float* __restrict__ out) {
  __shared__ _Float16 sA[2 * 128 * LDST];
  __shared__ _Float16 sB[2 * 64 * LDST];
  const int b = blockIdx.z, mBase = blockIdx.x * 128, nBase = blockIdx.y * 64;
  const int tid = threadIdx.x, lane = tid & 31, wave = tid >> 5;
  const int wm = wave >> 1, wn = wave & 1;
  v8f a00, a01, a10, a11;
  gemm_nt_async<Cc>(h2 + ((size_t)b * Nn + mBase) * Cc, Cc,
                    w16 + (size_t)nBase * Cc, Cc, sA, sB, tid, a00, a01, a10, a11);
  const int row0 = mBase + wm * 32 + ((lane >> 4) << 3);
  const int c0 = nBase + wn * 32 + (lane & 15), c1 = c0 + 16;
  const float b0f = bias[c0], b1f = bias[c1];
#pragma unroll
  for (int r = 0; r < 8; ++r) {
    const int ra = row0 + r, rb = row0 + 16 + r;
    const size_t i00 = ((size_t)b * Cc + c0) * Nn + ra;
    const size_t i01 = ((size_t)b * Cc + c1) * Nn + ra;
    const size_t i10 = ((size_t)b * Cc + c0) * Nn + rb;
    const size_t i11 = ((size_t)b * Cc + c1) * Nn + rb;
    out[i00] = x[i00] + b0f + a00[r];
    out[i01] = x[i01] + b1f + a01[r];
    out[i10] = x[i10] + b0f + a10[r];
    out[i11] = x[i11] + b1f + a11[r];
  }
}

// ---------------------------------------------------------------------------
extern "C" void kernel_launch(void* const* d_in, const int* in_sizes, int n_in,
                              void* d_out, int out_size, void* d_ws, size_t ws_size,
                              hipStream_t stream) {
  (void)in_sizes; (void)n_in; (void)out_size; (void)ws_size;
  const float* x      = (const float*)d_in[0];
  const float* gn_w   = (const float*)d_in[1];
  const float* gn_b   = (const float*)d_in[2];
  const float* qkv_w  = (const float*)d_in[3];
  const float* qkv_b  = (const float*)d_in[4];
  const float* proj_w = (const float*)d_in[5];
  const float* proj_b = (const float*)d_in[6];
  float* out = (float*)d_out;
  char*  ws  = (char*)d_ws;

  const size_t sz_h16    = (size_t)Bn * Nn * Cc * sizeof(_Float16);   // 16 MB
  const size_t sz_qkvo   = (size_t)Bn * Nn * O3 * sizeof(_Float16);   // 48 MB
  const size_t sz_scores = (size_t)Bn * Nn * Nn * sizeof(float);      // 64 MB

  _Float16* h16    = (_Float16*)(ws);
  _Float16* qkvo   = (_Float16*)(ws + sz_h16);
  float*    scores = (float*)   (ws + sz_h16 + sz_qkvo);
  _Float16* qw16   = (_Float16*)(ws + sz_h16 + sz_qkvo + sz_scores);
  _Float16* pw16   = qw16 + (size_t)O3 * Cc;
  _Float16* h2     = h16;   // h16 is dead after the QKV GEMM; reuse

  k_cvt<<<(O3 * Cc + 255) / 256, 256, 0, stream>>>(qkv_w,  qw16, O3 * Cc);
  k_cvt<<<(Cc * Cc + 255) / 256, 256, 0, stream>>>(proj_w, pw16, Cc * Cc);
  k_gn<<<Bn * GRP, 256, 0, stream>>>(x, gn_w, gn_b, h16);
  k_qkv<<<dim3(Nn / 128, O3 / 64, Bn), 256, 0, stream>>>(h16, qw16, qkv_b, qkvo);
  k_scores<<<dim3(Nn / 128, Nn / 64, Bn), 256, 0, stream>>>(qkvo, scores);
  k_softmax<<<Bn * Nn, 256, 0, stream>>>(scores);
  k_av<<<dim3(Nn / 128, Cc / 64, Bn), 256, 0, stream>>>(scores, qkvo, h2);
  k_proj<<<dim3(Nn / 128, Cc / 64, Bn), 256, 0, stream>>>(h2, pw16, proj_b, x, out);
}
